// TokenGNN_67319317398053
// MI455X (gfx1250) — matrix operand order repaired and checked by reading, not compiled, not run on hardware
//
#include <hip/hip_runtime.h>
#include <hip/hip_bf16.h>

// ---------------------------------------------------------------------------
// GAT (2 layers + input projection, JumpingKnowledge concat) for MI455X gfx1250
// GEMMs via v_wmma_f32_16x16x32_bf16 with b128 LDS fragment loads and register
// double-buffered global staging. Edge softmax/aggregation via native f32
// global atomics. All scratch in d_ws; all launches on stream.
// ---------------------------------------------------------------------------

#define GN 100000
#define GE 300000
#define GD 512
#define GH 256
#define NEG_SLOPE 0.2f

typedef __attribute__((ext_vector_type(16))) __bf16 v16bf;
typedef __attribute__((ext_vector_type(8)))  float  v8f;

// ---------------- ordered-uint float max-atomic helpers ----------------
__device__ __forceinline__ unsigned f2ord(float f) {
    unsigned b = __float_as_uint(f);
    return (b & 0x80000000u) ? ~b : (b | 0x80000000u);
}
__device__ __forceinline__ float ord2f(unsigned u) {
    unsigned b = (u & 0x80000000u) ? (u & 0x7FFFFFFFu) : ~u;
    return __uint_as_float(b);
}

// ---------------- WMMA bf16 GEMM: C[M,N] = A[M,K] @ B[K,N] (+bias) ----------
// B stride is hardcoded to GH (true for every call site) so the strided B
// staging loads use one base pointer + immediate offsets. A-tail rows are
// clamped (not predicated): garbage rows are computed but never stored.
#define BM 128
#define BN 64
#define BK 32
#define APITCH 40   // bf16 elems per LDS row (80 B: 16B aligned, bank-stride 20)

__global__ __launch_bounds__(256)
void gemm_bf16_wmma(const float* __restrict__ A, int lda,
                    const float* __restrict__ B,        // ldb == GH
                    float* __restrict__ C, int ldc,
                    const float* __restrict__ bias,
                    int M, int K)
{
    __shared__ alignas(16) __bf16 sA [BM][APITCH];   // A tile  [row][k]
    __shared__ alignas(16) __bf16 sBt[BN][APITCH];   // B tile transposed [col][k]

    const int tid  = threadIdx.x;
    const int wave = tid >> 5;
    const int lane = tid & 31;
    const int half = lane >> 4;
    const int l16  = lane & 15;

    const int block_m = blockIdx.x * BM;
    const int block_n = blockIdx.y * BN;

    // A staging: each thread owns 16 consecutive floats of one row.
    // Clamp the row: out-of-range rows load valid memory, results discarded.
    const int a_row = tid >> 1;
    const int a_kq  = (tid & 1) * 16;
    const int a_gr  = min(block_m + a_row, M - 1);
    const float* a_base = A + (size_t)a_gr * lda + a_kq;

    // B staging: each thread owns 8 floats along K at a fixed column
    const int b_col = tid & 63;
    const int b_kq  = (tid >> 6) * 8;
    const float* b_base = B + (size_t)b_kq * GH + block_n + b_col;

    float a_reg[16];
    float b_reg[8];

    // prologue: fetch tile k0 = 0 into registers
#pragma unroll
    for (int j = 0; j < 16; ++j) a_reg[j] = a_base[j];
#pragma unroll
    for (int j = 0; j < 8; ++j)  b_reg[j] = b_base[j * GH];

    v8f acc[4] = {};

    for (int k0 = 0; k0 < K; k0 += BK) {
        // ---- stage registers -> LDS as bf16, vectorized b128 stores ----
        {
            union { __bf16 h[16]; uint4 u[2]; } t;
#pragma unroll
            for (int j = 0; j < 16; ++j) t.h[j] = (__bf16)a_reg[j];
            *reinterpret_cast<uint4*>(&sA[a_row][a_kq])     = t.u[0];
            *reinterpret_cast<uint4*>(&sA[a_row][a_kq + 8]) = t.u[1];
        }
        {
            union { __bf16 h[8]; uint4 u; } t;
#pragma unroll
            for (int j = 0; j < 8; ++j) t.h[j] = (__bf16)b_reg[j];
            *reinterpret_cast<uint4*>(&sBt[b_col][b_kq]) = t.u;
        }
        __syncthreads();

        // ---- prefetch next tile into registers (overlaps the WMMAs) ----
        const int kn = k0 + BK;
        if (kn < K) {
            const float* ap = a_base + kn;
#pragma unroll
            for (int j = 0; j < 16; ++j) a_reg[j] = ap[j];
            const float* bp = b_base + (size_t)kn * GH;
#pragma unroll
            for (int j = 0; j < 8; ++j)  b_reg[j] = bp[j * GH];
        }

        // ---- A fragment: lanes 0-15 rows M, K = {half*8..+7, 16+half*8..+7}
        union { uint4 u[2]; v16bf v; } afu;
        const int arow = wave * 16 + l16;
        afu.u[0] = *reinterpret_cast<const uint4*>(&sA[arow][half * 8]);
        afu.u[1] = *reinterpret_cast<const uint4*>(&sA[arow][16 + half * 8]);

#pragma unroll
        for (int t = 0; t < 4; ++t) {
            // B fragment: col = t*16 + l16, K = {half*8..+7, 16+half*8..+7}
            union { uint4 u[2]; v16bf v; } bfu;
            const int bcol = t * 16 + l16;
            bfu.u[0] = *reinterpret_cast<const uint4*>(&sBt[bcol][half * 8]);
            bfu.u[1] = *reinterpret_cast<const uint4*>(&sBt[bcol][16 + half * 8]);
            acc[t] = __builtin_amdgcn_wmma_f32_16x16x32_bf16(
                false, afu.v, false, bfu.v, (short)0, acc[t], false, false);
        }
        __syncthreads();
    }

    // store: C/D 16x16 f32 layout: VGPR r -> M = r + half*8, N = lane&15
    const int row_base = block_m + wave * 16 + half * 8;
#pragma unroll
    for (int t = 0; t < 4; ++t) {
        const int col = block_n + t * 16 + l16;
        const float bv = bias ? bias[col] : 0.0f;
#pragma unroll
        for (int r = 0; r < 8; ++r) {
            const int gr = row_base + r;
            if (gr < M)
                C[(size_t)gr * ldc + col] = acc[t][r] + bv;
        }
    }
}

// ---------------- alpha_s / alpha_d: wave-per-node dot products -------------
__global__ __launch_bounds__(256)
void alpha_kernel(const float* __restrict__ hp,
                  const float* __restrict__ a_src,
                  const float* __restrict__ a_dst,
                  float* __restrict__ as, float* __restrict__ ad, int n)
{
    const int node = blockIdx.x * 8 + (threadIdx.x >> 5);
    const int lane = threadIdx.x & 31;
    if (node >= n) return;
    const float* row = hp + (size_t)node * GH;
    float s = 0.0f, d = 0.0f;
#pragma unroll
    for (int j = 0; j < GH / 32; ++j) {
        const int c = lane + 32 * j;
        const float v = row[c];
        s += v * a_src[c];
        d += v * a_dst[c];
    }
#pragma unroll
    for (int off = 16; off > 0; off >>= 1) {
        s += __shfl_xor(s, off, 32);
        d += __shfl_xor(d, off, 32);
    }
    if (lane == 0) { as[node] = s; ad[node] = d; }
}

// ---------------- edge phase ------------------------------------------------
__device__ __forceinline__ void edge_endpoints(const int* ei, int i,
                                               int& s, int& t)
{
    if (i < GE) { s = ei[i]; t = ei[GE + i]; }
    else        { s = t = i - GE; }          // self loops appended
}

__global__ __launch_bounds__(256)
void edge_logits(const int* __restrict__ ei,
                 const float* __restrict__ as, const float* __restrict__ ad,
                 float* __restrict__ ew, unsigned* __restrict__ mord)
{
    const int i = blockIdx.x * 256 + threadIdx.x;
    if (i >= GE + GN) return;
    int s, t; edge_endpoints(ei, i, s, t);
    float e = as[s] + ad[t];
    e = (e >= 0.0f) ? e : NEG_SLOPE * e;     // leaky relu
    ew[i] = e;
    atomicMax(mord + t, f2ord(e));
}

__global__ __launch_bounds__(256)
void edge_exp(const int* __restrict__ ei,
              float* __restrict__ ew,
              const unsigned* __restrict__ mord,
              float* __restrict__ denom)
{
    const int i = blockIdx.x * 256 + threadIdx.x;
    if (i >= GE + GN) return;
    int s, t; edge_endpoints(ei, i, s, t);
    (void)s;
    const float m = ord2f(mord[t]);
    const float w = __expf(ew[i] - m);
    ew[i] = w;
    atomicAdd(denom + t, w);
}

// wave-per-edge weighted scatter: acc[dst] += hp[src] * (w / denom[dst])
__global__ __launch_bounds__(256)
void edge_scatter(const int* __restrict__ ei,
                  const float* __restrict__ ew,
                  const float* __restrict__ denom,
                  const float* __restrict__ hp,
                  float* __restrict__ acc)
{
    const int edge = blockIdx.x * 8 + (threadIdx.x >> 5);
    const int lane = threadIdx.x & 31;
    if (edge >= GE + GN) return;
    int s, t; edge_endpoints(ei, edge, s, t);
    const float c = ew[edge] / denom[t];
    const float* hrow = hp  + (size_t)s * GH;
    float*       arow = acc + (size_t)t * GH;
#pragma unroll
    for (int j = 0; j < GH / 32; ++j) {
        const int col = lane + 32 * j;
        atomicAdd(arow + col, hrow[col] * c);
    }
}

// ---------------- finalize: out[:, off:off+H] = relu(acc + b) ---------------
__global__ __launch_bounds__(256)
void finalize_node(const float* __restrict__ acc,
                   const float* __restrict__ bias,
                   float* __restrict__ out, int coloff)
{
    const size_t idx = (size_t)blockIdx.x * 256 + threadIdx.x;
    if (idx >= (size_t)GN * GH) return;
    const int row = (int)(idx >> 8);
    const int c   = (int)(idx & (GH - 1));
    const float v = acc[idx] + bias[c];
    out[(size_t)row * (3 * GH) + coloff + c] = fmaxf(v, 0.0f);
}

// ---------------------------------------------------------------------------
extern "C" void kernel_launch(void* const* d_in, const int* in_sizes, int n_in,
                              void* d_out, int out_size, void* d_ws, size_t ws_size,
                              hipStream_t stream)
{
    (void)in_sizes; (void)n_in; (void)out_size; (void)ws_size;

    const float* x      = (const float*)d_in[0];
    const float* W_in   = (const float*)d_in[1];
    const float* b_in   = (const float*)d_in[2];
    const float* W1     = (const float*)d_in[3];
    const float* a_src1 = (const float*)d_in[4];
    const float* a_dst1 = (const float*)d_in[5];
    const float* b1     = (const float*)d_in[6];
    const float* W2     = (const float*)d_in[7];
    const float* a_src2 = (const float*)d_in[8];
    const float* a_dst2 = (const float*)d_in[9];
    const float* b2     = (const float*)d_in[10];
    const int*   ei     = (const int*)d_in[11];

    float* out = (float*)d_out;                 // [N, 3H]

    // workspace layout
    float*    hp    = (float*)d_ws;             // N*H
    float*    acc   = hp + (size_t)GN * GH;     // N*H
    float*    as    = acc + (size_t)GN * GH;    // N
    float*    ad    = as + GN;                  // N
    unsigned* mord  = (unsigned*)(ad + GN);     // N
    float*    denom = (float*)(mord + GN);      // N
    float*    ew    = denom + GN;               // E+N

    const dim3 blk(256);
    const dim3 gemm_grid((GN + BM - 1) / BM, GH / BN);
    const int  edges      = GE + GN;
    const int  grid_edge  = (edges + 255) / 256;
    const int  grid_wedge = (edges + 7) / 8;
    const int  grid_node  = (GN + 7) / 8;
    const int  grid_fin   = (int)(((size_t)GN * GH + 255) / 256);

    // ---- input projection: h0 = x @ W_in + b_in  -> out cols [0,256) ----
    gemm_bf16_wmma<<<gemm_grid, blk, 0, stream>>>(x, GD, W_in,
                                                  out, 3 * GH, b_in, GN, GD);

    // ---- GAT layer 1 ----
    gemm_bf16_wmma<<<gemm_grid, blk, 0, stream>>>(out + 0, 3 * GH, W1,
                                                  hp, GH, nullptr, GN, GH);
    alpha_kernel<<<grid_node, blk, 0, stream>>>(hp, a_src1, a_dst1, as, ad, GN);
    hipMemsetAsync(mord,  0, (size_t)GN * sizeof(unsigned), stream);
    hipMemsetAsync(denom, 0, (size_t)GN * sizeof(float),    stream);
    hipMemsetAsync(acc,   0, (size_t)GN * GH * sizeof(float), stream);
    edge_logits <<<grid_edge,  blk, 0, stream>>>(ei, as, ad, ew, mord);
    edge_exp    <<<grid_edge,  blk, 0, stream>>>(ei, ew, mord, denom);
    edge_scatter<<<grid_wedge, blk, 0, stream>>>(ei, ew, denom, hp, acc);
    finalize_node<<<grid_fin,  blk, 0, stream>>>(acc, b1, out, GH);

    // ---- GAT layer 2 ----
    gemm_bf16_wmma<<<gemm_grid, blk, 0, stream>>>(out + GH, 3 * GH, W2,
                                                  hp, GH, nullptr, GN, GH);
    alpha_kernel<<<grid_node, blk, 0, stream>>>(hp, a_src2, a_dst2, as, ad, GN);
    hipMemsetAsync(mord,  0, (size_t)GN * sizeof(unsigned), stream);
    hipMemsetAsync(denom, 0, (size_t)GN * sizeof(float),    stream);
    hipMemsetAsync(acc,   0, (size_t)GN * GH * sizeof(float), stream);
    edge_logits <<<grid_edge,  blk, 0, stream>>>(ei, as, ad, ew, mord);
    edge_exp    <<<grid_edge,  blk, 0, stream>>>(ei, ew, mord, denom);
    edge_scatter<<<grid_wedge, blk, 0, stream>>>(ei, ew, denom, hp, acc);
    finalize_node<<<grid_fin,  blk, 0, stream>>>(acc, b2, out, 2 * GH);
}
